// RGAT_24747601560018
// MI455X (gfx1250) — matrix operand order repaired and checked by reading, not compiled, not run on hardware
//
#include <hip/hip_runtime.h>

// ---------------- problem constants (match reference) ----------------
#define NN 50000
#define EE 400000
#define RR 3
#define INF_ 128
#define HIDF 64
#define CF 40
#define NH 3
#define NEG_SLOPE 0.2f

// ---------------- WMMA types ----------------
typedef __attribute__((ext_vector_type(16))) __bf16 v16bf;
typedef __attribute__((ext_vector_type(8)))  float  v8f;
typedef __attribute__((ext_vector_type(8)))  unsigned int v8u;

union ABfrag { v8u u; v16bf bf; };

__device__ __forceinline__ unsigned short f2bf(float f) {
    unsigned u = __float_as_uint(f);
    unsigned r = (u + 0x7FFFu + ((u >> 16) & 1u)) >> 16;
    return (unsigned short)r;
}

// ---------------- small utility kernels ----------------
__global__ void k_f2bf(const float* __restrict__ in, unsigned short* __restrict__ out, int n) {
    int i = blockIdx.x * blockDim.x + threadIdx.x;
    if (i < n) out[i] = f2bf(in[i]);
}

__global__ void k_fill(float* __restrict__ p, float v, int n) {
    int i = blockIdx.x * blockDim.x + threadIdx.x;
    if (i < n) p[i] = v;
}

// Pack W [K x KoutActual] f32 row-major into B-fragment layout:
// Bpack[((ntile*ksteps + ks)*32 + lane)*8 + dw] = pair (k, k+1) as 2 bf16,
// with k = ks*32 + (lane>>4)*16 + dw*2, n = ntile*16 + (lane&15). Zero-pad n >= KoutActual.
__global__ void k_packW(const float* __restrict__ W, unsigned int* __restrict__ Bp,
                        int K, int KoutActual, int ntiles, int ksteps) {
    int tid = blockIdx.x * blockDim.x + threadIdx.x;
    int total = ntiles * ksteps * 32 * 8;
    if (tid >= total) return;
    int dw = tid & 7;
    int lane = (tid >> 3) & 31;
    int t = tid >> 8;
    int ks = t % ksteps;
    int nt = t / ksteps;
    int k = ks * 32 + (lane >> 4) * 16 + dw * 2;
    int n = nt * 16 + (lane & 15);
    unsigned lo = 0, hi = 0;
    if (n < KoutActual) {
        lo = f2bf(W[(size_t)k * KoutActual + n]);
        hi = f2bf(W[(size_t)(k + 1) * KoutActual + n]);
    }
    Bp[tid] = lo | (hi << 16);
}

// ---------------- WMMA GEMM: Z[nrows x Kout] = A[nrows x K](bf16) * Wpack ----------------
// grid.x = ntiles (Kout tiles of 16), grid.y covers nrows/16 tiles, 4 waves/block.
__global__ void k_gemm_wmma(const unsigned short* __restrict__ A,
                            const unsigned int* __restrict__ Bp,
                            float* __restrict__ Z,
                            int nrows, int K, int Kout, int ksteps) {
    const int lane = threadIdx.x & 31;
    const int wv = threadIdx.x >> 5;
    const int mtiles = nrows >> 4;
    const int mt = blockIdx.y * 4 + wv;
    if (mt >= mtiles) return;                 // wave-uniform branch (EXEC all-ones inside)
    const int m0 = mt * 16;
    const int nt = blockIdx.x;
    const int half = lane >> 4;
    const int row = lane & 15;

    v8f acc = {};
    const unsigned short* arow = A + (size_t)(m0 + row) * K;
    for (int ks = 0; ks < ksteps; ++ks) {
        const int kbase = ks * 32 + half * 16;
        ABfrag a, b;
        const unsigned int* p0 = (const unsigned int*)(arow + kbase);
        const unsigned int* p1 = (const unsigned int*)(arow + kbase + 8);
        a.u[0] = p0[0]; a.u[1] = p0[1]; a.u[2] = p0[2]; a.u[3] = p0[3];
        a.u[4] = p1[0]; a.u[5] = p1[1]; a.u[6] = p1[2]; a.u[7] = p1[3];
        const unsigned int* bp = Bp + (((size_t)nt * ksteps + ks) * 32 + lane) * 8;
        #pragma unroll
        for (int i = 0; i < 8; ++i) b.u[i] = bp[i];
        acc = __builtin_amdgcn_wmma_f32_16x16x32_bf16(
            false, a.bf, false, b.bf, (short)0, acc, false, false);
    }
    const int n0 = nt * 16;
    const int col = n0 + row;
    if (col < Kout) {
        #pragma unroll
        for (int r = 0; r < 8; ++r) {
            int mrow = m0 + r + 8 * half;
            Z[(size_t)mrow * Kout + col] = acc[r];
        }
    }
}

// ---------------- attention kernels ----------------
// el[n,h] = sum_o z[n, h*out+o]*al[h,o]; er analog.
__global__ void k_elr(const float* __restrict__ Z, const float* __restrict__ al,
                      const float* __restrict__ ar, float* __restrict__ el,
                      float* __restrict__ er, int n, int out) {
    int i = blockIdx.x * blockDim.x + threadIdx.x;
    if (i >= n * NH) return;
    int node = i / NH, h = i - node * NH;
    const float* zp = Z + (size_t)node * NH * out + h * out;
    const float* alp = al + h * out;
    const float* arp = ar + h * out;
    float sl = 0.f, sr = 0.f;
    for (int o = 0; o < out; ++o) { float zv = zp[o]; sl += zv * alp[o]; sr += zv * arp[o]; }
    el[i] = sl; er[i] = sr;
}

__device__ __forceinline__ void atomicMaxF(float* addr, float v) {
    if (v >= 0.f) atomicMax((int*)addr, __float_as_int(v));
    else          atomicMin((unsigned int*)addr, __float_as_uint(v));
}

__global__ void k_edge_max(const int* __restrict__ src, const int* __restrict__ dst,
                           const float* __restrict__ el, const float* __restrict__ er,
                           float* __restrict__ ebuf, float* __restrict__ mmax) {
    int i = blockIdx.x * blockDim.x + threadIdx.x;
    if (i >= EE * NH) return;
    int e = i / NH, h = i - e * NH;
    int s = src[e], d = dst[e];
    float v = el[s * NH + h] + er[d * NH + h];
    v = (v > 0.f) ? v : NEG_SLOPE * v;       // leaky relu
    ebuf[i] = v;
    atomicMaxF(&mmax[d * NH + h], v);
}

__global__ void k_edge_sum(const int* __restrict__ dst, float* __restrict__ ebuf,
                           const float* __restrict__ mmax, float* __restrict__ ssum) {
    int i = blockIdx.x * blockDim.x + threadIdx.x;
    if (i >= EE * NH) return;
    int e = i / NH, h = i - e * NH;
    int d = dst[e];
    float a = __expf(ebuf[i] - mmax[d * NH + h]);
    ebuf[i] = a;
    atomicAdd(&ssum[d * NH + h], a);
}

// thread per (edge, channel c in [0, NH*out)): rst[dst, c] += alpha * z[src, c]
__global__ void k_edge_scatter(const int* __restrict__ src, const int* __restrict__ dst,
                               const float* __restrict__ ebuf, const float* __restrict__ ssum,
                               const float* __restrict__ Z, float* __restrict__ rst, int out) {
    const int stride = NH * out;
    long long i = (long long)blockIdx.x * blockDim.x + threadIdx.x;
    if (i >= (long long)EE * stride) return;
    int e = (int)(i / stride);
    int c = (int)(i - (long long)e * stride);
    int h = c / out;
    int s = src[e], d = dst[e];
    float alpha = ebuf[e * NH + h] / ssum[d * NH + h];
    atomicAdd(&rst[(size_t)d * stride + c], alpha * Z[(size_t)s * stride + c]);
}

// accum[n,o] += scale * sum_h act(rst[n, h*out+o] + b[h*out+o])
__global__ void k_epilogue(const float* __restrict__ rst, const float* __restrict__ b,
                           float* __restrict__ accum, int n, int out, int relu, float scale) {
    int i = blockIdx.x * blockDim.x + threadIdx.x;
    if (i >= n * out) return;
    int node = i / out, o = i - node * out;
    float acc = 0.f;
    for (int h = 0; h < NH; ++h) {
        float v = rst[(size_t)node * NH * out + h * out + o] + b[h * out + o];
        if (relu) v = v > 0.f ? v : 0.f;
        acc += v;
    }
    accum[i] += scale * acc;
}

// ---------------- host orchestration ----------------
static inline int cdiv(long long a, int b) { return (int)((a + b - 1) / b); }

extern "C" void kernel_launch(void* const* d_in, const int* in_sizes, int n_in,
                              void* d_out, int out_size, void* d_ws, size_t ws_size,
                              hipStream_t stream) {
    (void)in_sizes; (void)n_in; (void)out_size; (void)ws_size;
    const float* feat = (const float*)d_in[0];
    const int*   src  = (const int*)d_in[1];
    const int*   dst  = (const int*)d_in[2];
    const float* W1   = (const float*)d_in[3];
    const float* al1  = (const float*)d_in[4];
    const float* ar1  = (const float*)d_in[5];
    const float* b1   = (const float*)d_in[6];
    const float* W2   = (const float*)d_in[7];
    const float* al2  = (const float*)d_in[8];
    const float* ar2  = (const float*)d_in[9];
    const float* b2   = (const float*)d_in[10];
    float* out = (float*)d_out;

    // workspace carve-up
    char* ws = (char*)d_ws;
    size_t off = 0;
    auto carve = [&](size_t bytes) { char* p = ws + off; off = (off + bytes + 255) & ~(size_t)255; return p; };
    unsigned short* featb = (unsigned short*)carve((size_t)NN * INF_ * 2);
    float*          h1    = (float*)carve((size_t)NN * HIDF * 4);
    unsigned short* h1b   = (unsigned short*)carve((size_t)NN * HIDF * 2);
    unsigned int*   wpack = (unsigned int*)carve(64 * 1024);
    float*          z     = (float*)carve((size_t)NN * NH * HIDF * 4);   // max stride 192
    float*          rst   = (float*)carve((size_t)NN * NH * HIDF * 4);
    float*          el    = (float*)carve((size_t)NN * NH * 4);
    float*          er    = (float*)carve((size_t)NN * NH * 4);
    float*          mmax  = (float*)carve((size_t)NN * NH * 4);
    float*          ssum  = (float*)carve((size_t)NN * NH * 4);
    float*          ebuf  = (float*)carve((size_t)EE * NH * 4);

    const int BLK = 256;
    const float NEG_INF = -__builtin_huge_valf();

    // feat -> bf16
    k_f2bf<<<cdiv((long long)NN * INF_, BLK), BLK, 0, stream>>>(feat, featb, NN * INF_);
    k_fill<<<cdiv((long long)NN * HIDF, BLK), BLK, 0, stream>>>(h1, 0.f, NN * HIDF);

    // ---------------- layer 1: IN=128 -> H*HID=192, relu ----------------
    {
        const int K = INF_, KO = NH * HIDF;          // 128, 192
        const int ntiles = (KO + 15) / 16, ksteps = K / 32;  // 12, 4
        dim3 ggrid(ntiles, cdiv(NN / 16, 4));
        for (int r = 0; r < RR; ++r) {
            k_packW<<<cdiv((long long)ntiles * ksteps * 256, BLK), BLK, 0, stream>>>(
                W1 + (size_t)r * K * KO, wpack, K, KO, ntiles, ksteps);
            k_gemm_wmma<<<ggrid, 128, 0, stream>>>(featb, wpack, z, NN, K, KO, ksteps);
            k_elr<<<cdiv((long long)NN * NH, BLK), BLK, 0, stream>>>(
                z, al1 + (size_t)r * NH * HIDF, ar1 + (size_t)r * NH * HIDF, el, er, NN, HIDF);
            k_fill<<<cdiv((long long)NN * NH, BLK), BLK, 0, stream>>>(mmax, NEG_INF, NN * NH);
            k_fill<<<cdiv((long long)NN * NH, BLK), BLK, 0, stream>>>(ssum, 0.f, NN * NH);
            k_fill<<<cdiv((long long)NN * KO, BLK), BLK, 0, stream>>>(rst, 0.f, NN * KO);
            k_edge_max<<<cdiv((long long)EE * NH, BLK), BLK, 0, stream>>>(
                src + (size_t)r * EE, dst + (size_t)r * EE, el, er, ebuf, mmax);
            k_edge_sum<<<cdiv((long long)EE * NH, BLK), BLK, 0, stream>>>(
                dst + (size_t)r * EE, ebuf, mmax, ssum);
            k_edge_scatter<<<cdiv((long long)EE * KO, BLK), BLK, 0, stream>>>(
                src + (size_t)r * EE, dst + (size_t)r * EE, ebuf, ssum, z, rst, HIDF);
            k_epilogue<<<cdiv((long long)NN * HIDF, BLK), BLK, 0, stream>>>(
                rst, b1 + (size_t)r * KO, h1, NN, HIDF, 1, 1.f / (NH * RR));
        }
    }

    // h1 -> bf16
    k_f2bf<<<cdiv((long long)NN * HIDF, BLK), BLK, 0, stream>>>(h1, h1b, NN * HIDF);
    k_fill<<<cdiv((long long)NN * CF, BLK), BLK, 0, stream>>>(out, 0.f, NN * CF);

    // ---------------- layer 2: HID=64 -> H*C=120, no activation ----------------
    {
        const int K = HIDF, KO = NH * CF;            // 64, 120
        const int ntiles = (KO + 15) / 16, ksteps = K / 32;  // 8 (padded), 2
        dim3 ggrid(ntiles, cdiv(NN / 16, 4));
        for (int r = 0; r < RR; ++r) {
            k_packW<<<cdiv((long long)ntiles * ksteps * 256, BLK), BLK, 0, stream>>>(
                W2 + (size_t)r * K * KO, wpack, K, KO, ntiles, ksteps);
            k_gemm_wmma<<<ggrid, 128, 0, stream>>>(h1b, wpack, z, NN, K, KO, ksteps);
            k_elr<<<cdiv((long long)NN * NH, BLK), BLK, 0, stream>>>(
                z, al2 + (size_t)r * NH * CF, ar2 + (size_t)r * NH * CF, el, er, NN, CF);
            k_fill<<<cdiv((long long)NN * NH, BLK), BLK, 0, stream>>>(mmax, NEG_INF, NN * NH);
            k_fill<<<cdiv((long long)NN * NH, BLK), BLK, 0, stream>>>(ssum, 0.f, NN * NH);
            k_fill<<<cdiv((long long)NN * KO, BLK), BLK, 0, stream>>>(rst, 0.f, NN * KO);
            k_edge_max<<<cdiv((long long)EE * NH, BLK), BLK, 0, stream>>>(
                src + (size_t)r * EE, dst + (size_t)r * EE, el, er, ebuf, mmax);
            k_edge_sum<<<cdiv((long long)EE * NH, BLK), BLK, 0, stream>>>(
                dst + (size_t)r * EE, ebuf, mmax, ssum);
            k_edge_scatter<<<cdiv((long long)EE * KO, BLK), BLK, 0, stream>>>(
                src + (size_t)r * EE, dst + (size_t)r * EE, ebuf, ssum, z, rst, CF);
            k_epilogue<<<cdiv((long long)NN * CF, BLK), BLK, 0, stream>>>(
                rst, b2 + (size_t)r * KO, out, NN, CF, 0, 1.f / (NH * RR));
        }
    }
}
